// MultiAttention_1889785610927
// MI455X (gfx1250) — matrix operand order repaired and checked by reading, not compiled, not run on hardware
//
#include <hip/hip_runtime.h>
#include <hip/hip_bf16.h>

// ---------------------------------------------------------------------------
// MultiAttention for MI455X (gfx1250, wave32, WMMA 16x16x32 bf16)
//   x:  [B,S,DIN]  f32      W: [H,3,DIN,E] f32     Wo: [H*E, E] f32
//   out:[B,S,E]    f32
// B=16 S=1024 DIN=1024 H=16 E=128
// LDS operand tiles laid out so every WMMA fragment is two ds_load_b128 ops
// (A row-major, B transposed/N-major).  Global->LDS staging is software-
// pipelined through registers so global loads overlap the WMMA phase.
// ---------------------------------------------------------------------------

typedef __bf16 bf16_t;
typedef __attribute__((ext_vector_type(16))) __bf16 v16bf;
typedef __attribute__((ext_vector_type(8)))  float  v8f;
typedef __attribute__((ext_vector_type(4)))  float  v4f;
typedef __attribute__((ext_vector_type(4)))  __bf16 v4bf;

#define B_   16
#define S_   1024
#define DIN_ 1024
#define H_   16
#define E_   128
#define F_   (H_ * E_)   // 2048

// ---------------------------------------------------------------------------
// WMMA helpers (contiguous k order: lane half h holds k = 16h..16h+15)
// ---------------------------------------------------------------------------
__device__ __forceinline__ v8f wmma_bf16(v16bf a, v16bf b, v8f c) {
  return __builtin_amdgcn_wmma_f32_16x16x32_bf16(false, a, false, b,
                                                 (short)0, c, false, false);
}

// A 16(M)x32(K): lane&15 = M; lanes 0-15 hold K 0-15, lanes 16-31 hold
// K 16-31 -> two contiguous 16-byte chunks per lane.
__device__ __forceinline__ v16bf load_a_frag(const bf16_t* p, int ld) {
  const int lane = threadIdx.x & 31;
  const int half = lane >> 4;
  const int m    = lane & 15;
  v16bf a;
#pragma unroll
  for (int v = 0; v < 8; ++v) {
    const int k = 2 * v + half * 16;
    a[2 * v]     = p[m * ld + k];
    a[2 * v + 1] = p[m * ld + k + 1];
  }
  return a;
}

// B 32(K)x16(N) read from a TRANSPOSED (N-major) tile Bt[n][k], row stride
// ld: lane&15 = N; lanes 0-15 hold K 0-15, lanes 16-31 hold K 16-31.
// Per-lane data = 16 contiguous bf16 (32B, 16B aligned) -> 2x ds_load_b128.
__device__ __forceinline__ v16bf load_b_frag_t(const bf16_t* p, int ld) {
  const int lane = threadIdx.x & 31;
  const int half = lane >> 4;
  const int n    = lane & 15;
  const bf16_t* q = p + n * ld + 16 * half;
  v16bf b;
#pragma unroll
  for (int j = 0; j < 16; ++j) b[j] = q[j];
  return b;
}

union U4 { uint4 u4; unsigned u[4]; bf16_t x[8]; };

// ---------------------------------------------------------------------------
// f32 -> bf16 conversion (vectorized x4)
// ---------------------------------------------------------------------------
__global__ void __launch_bounds__(256)
cvt_f32_to_bf16(const float* __restrict__ in, bf16_t* __restrict__ out, int n4) {
  const v4f* in4  = (const v4f*)in;
  v4bf*      out4 = (v4bf*)out;
  int stride = gridDim.x * 256;
  for (int i = blockIdx.x * 256 + threadIdx.x; i < n4; i += stride) {
    v4f v = in4[i];
    v4bf o;
    o[0] = (bf16_t)v[0]; o[1] = (bf16_t)v[1];
    o[2] = (bf16_t)v[2]; o[3] = (bf16_t)v[3];
    out4[i] = o;
  }
}

// ---------------------------------------------------------------------------
// QKV projection: for z=(b,h,c): [S x DIN] @ [DIN x E] -> bf16 [B,H,S,E]
// grid: x = S/16, z = B*H*3.  128 threads = 4 waves; each wave owns a
// 16x32 slice of C (two accumulators), block covers all E=128 columns.
// Staging is register-pipelined: tile i+1 global loads issue while tile i
// is consumed by the WMMAs.
// ---------------------------------------------------------------------------
__global__ void __launch_bounds__(128)
qkv_gemm(const bf16_t* __restrict__ xb, const bf16_t* __restrict__ wb,
         bf16_t* __restrict__ qb, bf16_t* __restrict__ kb,
         bf16_t* __restrict__ vb) {
  __shared__ bf16_t lds_a[16 * 32];      // A tile, row-major
  __shared__ bf16_t lds_bt[128 * 32];    // B tile transposed: [n][k]

  const int tid  = threadIdx.x;
  const int wave = tid >> 5;
  const int lane = tid & 31;
  const int half = lane >> 4;
  const int nn   = lane & 15;

  const int z  = blockIdx.z;
  const int c  = z % 3;
  const int h  = (z / 3) % H_;
  const int b  = z / (3 * H_);
  const int m0 = blockIdx.x * 16;

  const bf16_t* A  = xb + (size_t)b * S_ * DIN_ + (size_t)m0 * DIN_;
  const bf16_t* Bw = wb + (size_t)(h * 3 + c) * DIN_ * E_;

  // staging coordinates
  const int ar = tid >> 2, ac = tid & 3;          // A: tid < 64
  const int br = tid >> 3, bc = tid & 7;          // B: rows br and br+16

  U4 areg, breg0, breg1;
  auto stage_load = [&](int k0) {
    if (tid < 64)
      areg.u4 = *(const uint4*)(A + (size_t)ar * DIN_ + k0 + ac * 8);
    breg0.u4 = *(const uint4*)(Bw + (size_t)(k0 + br) * E_ + bc * 8);
    breg1.u4 = *(const uint4*)(Bw + (size_t)(k0 + br + 16) * E_ + bc * 8);
  };

  v8f acc0 = {}, acc1 = {};
  stage_load(0);

  for (int k0 = 0; k0 < DIN_; k0 += 32) {
    __syncthreads();   // previous tile fully consumed
    if (tid < 64) ((uint4*)lds_a)[tid] = areg.u4;
#pragma unroll
    for (int j = 0; j < 8; ++j) {
      lds_bt[(bc * 8 + j) * 32 + br]      = breg0.x[j];
      lds_bt[(bc * 8 + j) * 32 + br + 16] = breg1.x[j];
    }
    __syncthreads();   // tile visible
    if (k0 + 32 < DIN_) {
      stage_load(k0 + 32);   // overlaps WMMA phase
      if (k0 + 64 < DIN_)
        __builtin_prefetch((const void*)(Bw + (size_t)(k0 + 64 + br) * E_), 0, 0);
    }

    v16bf af = load_a_frag(lds_a, 32);
    acc0 = wmma_bf16(af, load_b_frag_t(lds_bt + (wave * 32) * 32, 32), acc0);
    acc1 = wmma_bf16(af, load_b_frag_t(lds_bt + (wave * 32 + 16) * 32, 32), acc1);
  }

  bf16_t* dstbase = (c == 0 ? qb : (c == 1 ? kb : vb));
  bf16_t* dst = dstbase + ((size_t)(b * H_ + h) * S_ + m0) * E_ + wave * 32;
#pragma unroll
  for (int r = 0; r < 8; ++r) {
    dst[(size_t)(r + 8 * half) * E_ + nn]      = (bf16_t)acc0[r];
    dst[(size_t)(r + 8 * half) * E_ + 16 + nn] = (bf16_t)acc1[r];
  }
}

// ---------------------------------------------------------------------------
// Flash attention per (b,h).  grid: x = S/64, y = B*H.  128 threads = 4
// waves, each wave owns 16 query rows; loop over key tiles of 32.
// scores = Q K^T / 64, online softmax, O = P V -> bf16 [B,S,H*E]
// K tile row-major IS the transposed-B layout for scores; V is staged
// transposed for the P.V GEMM.  K/V staging register-pipelined.
// ---------------------------------------------------------------------------
__global__ void __launch_bounds__(128)
attn_kernel(const bf16_t* __restrict__ qb, const bf16_t* __restrict__ kb,
            const bf16_t* __restrict__ vb, bf16_t* __restrict__ ob) {
  __shared__ bf16_t lds_q[64 * 128];    // Q block, row-major
  __shared__ bf16_t lds_k[32 * 128];    // K tile row-major [t][e]
  __shared__ bf16_t lds_vt[128 * 32];   // V tile transposed [e][t]
  __shared__ bf16_t lds_p[4][16 * 32];  // per-wave P relayout buffer

  const int tid  = threadIdx.x;
  const int wave = tid >> 5;
  const int lane = tid & 31;
  const int half = lane >> 4;
  const int nn   = lane & 15;

  const int bh = blockIdx.y;
  const int b  = bh / H_;
  const int h  = bh % H_;
  const int s0 = blockIdx.x * 64;

  const size_t base = (size_t)bh * S_ * E_;
  const bf16_t* Q = qb + base + (size_t)s0 * E_;
  const bf16_t* K = kb + base;
  const bf16_t* V = vb + base;

  // Stage the 64x128 Q block (16 KB) once, coalesced.
  {
    const uint4* g = (const uint4*)Q;
    uint4*       l = (uint4*)lds_q;
#pragma unroll
    for (int i = tid; i < 1024; i += 128) l[i] = g[i];
  }
  __syncthreads();

  // Q fragments live in registers for the whole key loop.
  v16bf qf[4];
#pragma unroll
  for (int ke = 0; ke < 4; ++ke)
    qf[ke] = load_a_frag(lds_q + (size_t)(wave * 16) * 128 + ke * 32, 128);

  // staging coordinates (V transposed store)
  const int vt0 = tid >> 3,        ve0 = tid & 7;         // row t, e-octet
  const int vt1 = (tid >> 3) + 16, ve1 = tid & 7;

  U4 kreg[4], vreg0, vreg1;
  auto stage_load = [&](int t0) {
    const uint4* gk = (const uint4*)(K + (size_t)t0 * E_);
#pragma unroll
    for (int j = 0; j < 4; ++j) kreg[j].u4 = gk[tid + 128 * j];
    vreg0.u4 = *(const uint4*)(V + (size_t)(t0 + vt0) * E_ + ve0 * 8);
    vreg1.u4 = *(const uint4*)(V + (size_t)(t0 + vt1) * E_ + ve1 * 8);
  };

  v8f o[8];
#pragma unroll
  for (int et = 0; et < 8; ++et) o[et] = v8f{};
  float mrow[8], lrow[8];
#pragma unroll
  for (int r = 0; r < 8; ++r) { mrow[r] = -3.0e38f; lrow[r] = 0.0f; }

  const float scale = 1.0f / 64.0f;  // reference: / (E * 0.5)

  stage_load(0);

  for (int t0 = 0; t0 < S_; t0 += 32) {
    __syncthreads();   // previous K/V tile fully consumed
    {
      uint4* lk = (uint4*)lds_k;
#pragma unroll
      for (int j = 0; j < 4; ++j) lk[tid + 128 * j] = kreg[j].u4;
#pragma unroll
      for (int j = 0; j < 8; ++j) {
        lds_vt[(ve0 * 8 + j) * 32 + vt0] = vreg0.x[j];
        lds_vt[(ve1 * 8 + j) * 32 + vt1] = vreg1.x[j];
      }
    }
    __syncthreads();   // tile visible
    if (t0 + 32 < S_) stage_load(t0 + 32);   // overlaps score/PV WMMAs

    // Two 16-wide score tiles (t0..t0+15, t0+16..t0+31)
    v8f sa = {}, sb = {};
#pragma unroll
    for (int ke = 0; ke < 4; ++ke) {
      sa = wmma_bf16(qf[ke], load_b_frag_t(lds_k + ke * 32, 128), sa);
      sb = wmma_bf16(qf[ke], load_b_frag_t(lds_k + 16 * 128 + ke * 32, 128), sb);
    }

    float corr[8];
#pragma unroll
    for (int r = 0; r < 8; ++r) {
      float a  = sa[r] * scale;
      float bv = sb[r] * scale;
      // row max across the 16 lanes of this half-wave
      float mx = fmaxf(a, bv);
      mx = fmaxf(mx, __shfl_xor(mx, 1, 32));
      mx = fmaxf(mx, __shfl_xor(mx, 2, 32));
      mx = fmaxf(mx, __shfl_xor(mx, 4, 32));
      mx = fmaxf(mx, __shfl_xor(mx, 8, 32));
      const float mnew = fmaxf(mrow[r], mx);
      corr[r] = __expf(mrow[r] - mnew);
      const float p0 = __expf(a - mnew);
      const float p1 = __expf(bv - mnew);
      float rs = p0 + p1;
      rs += __shfl_xor(rs, 1, 32);
      rs += __shfl_xor(rs, 2, 32);
      rs += __shfl_xor(rs, 4, 32);
      rs += __shfl_xor(rs, 8, 32);
      lrow[r] = lrow[r] * corr[r] + rs;
      mrow[r] = mnew;
      // P relayout (C layout -> A layout) through per-wave LDS; same-wave
      // DS ops are in-order (ISA 7.3), no barrier needed.
      lds_p[wave][(r + 8 * half) * 32 + nn]      = (bf16_t)p0;
      lds_p[wave][(r + 8 * half) * 32 + 16 + nn] = (bf16_t)p1;
    }

    v16bf pf = load_a_frag(lds_p[wave], 32);
#pragma unroll
    for (int et = 0; et < 8; ++et) {
#pragma unroll
      for (int r = 0; r < 8; ++r) o[et][r] *= corr[r];
      o[et] = wmma_bf16(pf, load_b_frag_t(lds_vt + (et * 16) * 32, 32), o[et]);
    }
  }

  // Epilogue: normalize and store concat-head layout [b][s][h*e] as bf16.
  float invl[8];
#pragma unroll
  for (int r = 0; r < 8; ++r) invl[r] = 1.0f / lrow[r];

  bf16_t* dst = ob + (((size_t)b * S_ + s0 + wave * 16) * H_ + h) * E_;
#pragma unroll
  for (int et = 0; et < 8; ++et)
#pragma unroll
    for (int r = 0; r < 8; ++r)
      dst[(size_t)(r + 8 * half) * F_ + et * 16 + nn] =
          (bf16_t)(o[et][r] * invl[r]);
}

// ---------------------------------------------------------------------------
// Output projection: [B*S x F] @ [F x E] -> f32 out.
// grid: x = (B*S)/16.  128 threads = 4 waves; two 16x16 C tiles per wave
// (block covers all E=128 columns).  Register-pipelined staging.
// ---------------------------------------------------------------------------
__global__ void __launch_bounds__(128)
out_gemm(const bf16_t* __restrict__ ob, const bf16_t* __restrict__ wob,
         float* __restrict__ out) {
  __shared__ bf16_t lds_a[16 * 32];
  __shared__ bf16_t lds_bt[128 * 32];

  const int tid  = threadIdx.x;
  const int wave = tid >> 5;
  const int lane = tid & 31;
  const int half = lane >> 4;
  const int nn   = lane & 15;

  const int m0 = blockIdx.x * 16;
  const bf16_t* A = ob + (size_t)m0 * F_;

  const int ar = tid >> 2, ac = tid & 3;
  const int br = tid >> 3, bc = tid & 7;

  U4 areg, breg0, breg1;
  auto stage_load = [&](int k0) {
    if (tid < 64)
      areg.u4 = *(const uint4*)(A + (size_t)ar * F_ + k0 + ac * 8);
    breg0.u4 = *(const uint4*)(wob + (size_t)(k0 + br) * E_ + bc * 8);
    breg1.u4 = *(const uint4*)(wob + (size_t)(k0 + br + 16) * E_ + bc * 8);
  };

  v8f acc0 = {}, acc1 = {};
  stage_load(0);

  for (int k0 = 0; k0 < F_; k0 += 32) {
    __syncthreads();
    if (tid < 64) ((uint4*)lds_a)[tid] = areg.u4;
#pragma unroll
    for (int j = 0; j < 8; ++j) {
      lds_bt[(bc * 8 + j) * 32 + br]      = breg0.x[j];
      lds_bt[(bc * 8 + j) * 32 + br + 16] = breg1.x[j];
    }
    __syncthreads();
    if (k0 + 32 < F_) {
      stage_load(k0 + 32);
      if (k0 + 64 < F_)
        __builtin_prefetch((const void*)(A + (size_t)ar * F_ + k0 + 64), 0, 0);
    }

    v16bf af = load_a_frag(lds_a, 32);
    acc0 = wmma_bf16(af, load_b_frag_t(lds_bt + (wave * 32) * 32, 32), acc0);
    acc1 = wmma_bf16(af, load_b_frag_t(lds_bt + (wave * 32 + 16) * 32, 32), acc1);
  }

  float* dst = out + (size_t)m0 * E_ + wave * 32;
#pragma unroll
  for (int r = 0; r < 8; ++r) {
    dst[(size_t)(r + 8 * half) * E_ + nn]      = acc0[r];
    dst[(size_t)(r + 8 * half) * E_ + 16 + nn] = acc1[r];
  }
}

// ---------------------------------------------------------------------------
// Host launcher.  Workspace layout (bf16 buffers), ~315 MB total.
// ---------------------------------------------------------------------------
extern "C" void kernel_launch(void* const* d_in, const int* in_sizes, int n_in,
                              void* d_out, int out_size, void* d_ws,
                              size_t ws_size, hipStream_t stream) {
  (void)in_sizes; (void)n_in; (void)out_size; (void)ws_size;

  const float* x  = (const float*)d_in[0];
  const float* W  = (const float*)d_in[1];
  const float* Wo = (const float*)d_in[2];
  float* out = (float*)d_out;

  char* ws = (char*)d_ws;
  size_t off = 0;
  bf16_t* xb  = (bf16_t*)(ws + off); off += (size_t)B_ * S_ * DIN_ * 2;
  bf16_t* wb  = (bf16_t*)(ws + off); off += (size_t)H_ * 3 * DIN_ * E_ * 2;
  bf16_t* wob = (bf16_t*)(ws + off); off += (size_t)F_ * E_ * 2;
  bf16_t* qb  = (bf16_t*)(ws + off); off += (size_t)B_ * H_ * S_ * E_ * 2;
  bf16_t* kb  = (bf16_t*)(ws + off); off += (size_t)B_ * H_ * S_ * E_ * 2;
  bf16_t* vb  = (bf16_t*)(ws + off); off += (size_t)B_ * H_ * S_ * E_ * 2;
  bf16_t* ob  = (bf16_t*)(ws + off); off += (size_t)B_ * S_ * F_ * 2;

  // 1) f32 -> bf16 conversions
  {
    const int nx = B_ * S_ * DIN_ / 4;
    cvt_f32_to_bf16<<<dim3(2048), dim3(256), 0, stream>>>(x, xb, nx);
    const int nw = H_ * 3 * DIN_ * E_ / 4;
    cvt_f32_to_bf16<<<dim3(2048), dim3(256), 0, stream>>>(W, wb, nw);
    const int nwo = F_ * E_ / 4;
    cvt_f32_to_bf16<<<dim3(256), dim3(256), 0, stream>>>(Wo, wob, nwo);
  }

  // 2) QKV projection
  qkv_gemm<<<dim3(S_ / 16, 1, B_ * H_ * 3), dim3(128), 0, stream>>>(
      xb, wb, qb, kb, vb);

  // 3) Attention (flash-style, online softmax)
  attn_kernel<<<dim3(S_ / 64, B_ * H_), dim3(128), 0, stream>>>(qb, kb, vb, ob);

  // 4) Output projection
  out_gemm<<<dim3(B_ * S_ / 16, 1), dim3(128), 0, stream>>>(ob, wob, out);
}
